// MambaVAE_37658273252095
// MI455X (gfx1250) — matrix operand order repaired and checked by reading, not compiled, not run on hardware
//
#include <hip/hip_runtime.h>
#include <hip/hip_bf16.h>
#include <math.h>

// ---------------- problem constants ----------------
constexpr int H  = 768;
constexpr int I  = 1536;
constexpr int S  = 16;
constexpr int Kc = 4;       // conv width
constexpr int R  = 48;
constexpr int V  = 50280;
constexpr int NL = 12;
constexpr int B  = 4;
constexpr int L  = 128;
constexpr int M  = B * L;            // 512 token rows
constexpr int PJ = R + 2 * S;        // 80
constexpr int BIS   = B * I * S;     // 98304
constexpr int NLBIS = NL * BIS;      // 1179648

typedef __attribute__((ext_vector_type(16))) __bf16 v16bf;
typedef __attribute__((ext_vector_type(8)))  float  v8f;

static __device__ __forceinline__ __bf16 f2bf(float f) { return (__bf16)f; }

// =====================================================================
// GEMM: C[m,n] = epi( sum_k A[m,k] * W[n,k] )
//   A: M x K (row stride LDA), W: N x K (row stride LDB), C: M x N
//   EPI 0: plain   EPI 1: softplus(x + bias[n])   EPI 2: x + Res[m,n]
// bf16 WMMA 16x16x32, fp32 accumulate. Block = 256 thr = 8 waves;
// block tile 16(M) x 256(N); each wave owns two adjacent 16x16 tiles and
// reuses its A fragment across both v_wmma. All dims compile-time so the
// K-edge (K=48) masks fold; N-edge masking only exists when N % 256 != 0
// (x_proj N=80, logits N=50280). Loads are float4 (global_load_b128).
// EXEC is all-ones at every v_wmma (masking is done by value, not branch).
// =====================================================================
template<int N, int EPI>
static __device__ __forceinline__ void
epi_store(const v8f& acc, int m0, int half, int bn, bool ok,
          const float* __restrict__ bias, const float* __restrict__ Res,
          float* __restrict__ C)
{
  if (!ok) return;
  float bb = (EPI == 1) ? bias[bn] : 0.f;
  #pragma unroll
  for (int j = 0; j < 8; ++j) {
    int row = m0 + j + (half << 3);
    float v = acc[j];
    if (EPI == 1) { v += bb; v = (v > 20.f) ? v : log1pf(__expf(v)); }
    else if (EPI == 2) { v += Res[(size_t)row * N + bn]; }
    C[(size_t)row * N + bn] = v;
  }
}

template<int N, int K, int LDA, int LDB, int EPI>
__global__ void __launch_bounds__(256)
gemm_k(const float* __restrict__ A, const float* __restrict__ W,
       const float* __restrict__ bias, const float* __restrict__ Res,
       float* __restrict__ C)
{
  static_assert(K % 16 == 0, "K must be a multiple of 16");
  constexpr bool EDGE = (N % 256) != 0;              // N-edge can be hit?
  const int lane = threadIdx.x & 31;
  const int wave = threadIdx.x >> 5;                 // 0..7
  const int half = lane >> 4;                        // 0/1
  const int r    = lane & 15;
  const int m0   = blockIdx.y * 16;
  const int n0   = blockIdx.x * 256 + wave * 32;     // two tiles: n0, n0+16

  const float* Arow = A + (size_t)(m0 + r) * LDA;
  const int  bn0 = n0 + r,  bn1 = n0 + 16 + r;
  const bool ok0 = EDGE ? (bn0 < N) : true;
  const bool ok1 = EDGE ? (bn1 < N) : true;
  const float* W0 = W + (size_t)(ok0 ? bn0 : 0) * LDB;
  const float* W1 = W + (size_t)(ok1 ? bn1 : 0) * LDB;
  const float msk0 = ok0 ? 1.f : 0.f;
  const float msk1 = ok1 ? 1.f : 0.f;

  v8f acc0 = {}, acc1 = {};
  for (int k0 = 0; k0 < K; k0 += 32) {
    // ---- A fragment: two 8-float contiguous runs (float4 x2 each) ----
    float ar[16];
    #pragma unroll
    for (int g = 0; g < 2; ++g) {
      const int kb = k0 + half * 16 + g * 8;
      if (kb + 8 <= K) {                             // folds at compile time
        const float4* ap = (const float4*)(Arow + kb);
        float4 p0 = ap[0], p1 = ap[1];
        ar[g*8+0]=p0.x; ar[g*8+1]=p0.y; ar[g*8+2]=p0.z; ar[g*8+3]=p0.w;
        ar[g*8+4]=p1.x; ar[g*8+5]=p1.y; ar[g*8+6]=p1.z; ar[g*8+7]=p1.w;
      } else {
        #pragma unroll
        for (int j = 0; j < 8; ++j) ar[g*8+j] = 0.f;
      }
    }
    // 16-bit A 16x32 fragment, contiguous k order: a[e] <- k = 16*half + e (same as B)
    v16bf a;
    #pragma unroll
    for (int e = 0; e < 16; ++e) {
      int v = e >> 1;
      a[e] = f2bf(ar[((v >> 2) << 3) + ((v & 3) << 1) + (e & 1)]);
    }
    // ---- B fragments: 16 contiguous floats (float4 x4) per tile ----
    const int kb = k0 + half * 16;
    float br0[16], br1[16];
    if (kb + 16 <= K) {                              // folds at compile time
      const float4* bp0 = (const float4*)(W0 + kb);
      const float4* bp1 = (const float4*)(W1 + kb);
      #pragma unroll
      for (int q = 0; q < 4; ++q) {
        float4 t0 = bp0[q], t1 = bp1[q];
        if (EDGE) { t0.x*=msk0; t0.y*=msk0; t0.z*=msk0; t0.w*=msk0;
                    t1.x*=msk1; t1.y*=msk1; t1.z*=msk1; t1.w*=msk1; }
        br0[q*4+0]=t0.x; br0[q*4+1]=t0.y; br0[q*4+2]=t0.z; br0[q*4+3]=t0.w;
        br1[q*4+0]=t1.x; br1[q*4+1]=t1.y; br1[q*4+2]=t1.z; br1[q*4+3]=t1.w;
      }
    } else {
      #pragma unroll
      for (int e = 0; e < 16; ++e) { br0[e] = 0.f; br1[e] = 0.f; }
    }
    v16bf b0, b1;
    #pragma unroll
    for (int e = 0; e < 16; ++e) { b0[e] = f2bf(br0[e]); b1[e] = f2bf(br1[e]); }

    acc0 = __builtin_amdgcn_wmma_f32_16x16x32_bf16(false, a, false, b0,
                                                   (short)0, acc0, false, false);
    acc1 = __builtin_amdgcn_wmma_f32_16x16x32_bf16(false, a, false, b1,
                                                   (short)0, acc1, false, false);
  }

  epi_store<N, EPI>(acc0, m0, half, bn0, ok0, bias, Res, C);
  epi_store<N, EPI>(acc1, m0, half, bn1, ok1, bias, Res, C);
}

// ---------------- embedding gather ----------------
__global__ void __launch_bounds__(256)
embed_k(const int* __restrict__ ids, const float* __restrict__ emb, float* __restrict__ h)
{
  int t = blockIdx.x * 256 + threadIdx.x;            // < M*H
  int m = t / H, c = t % H;
  h[t] = emb[(size_t)ids[m] * H + c];
}

// ---------------- RMSNorm (one row per block) ----------------
__global__ void __launch_bounds__(256)
rmsnorm_k(const float* __restrict__ x, const float* __restrict__ w, float* __restrict__ o)
{
  __shared__ float red[256];
  int row = blockIdx.x, t = threadIdx.x;
  const float* xr = x + (size_t)row * H;
  float s = 0.f;
  for (int c = t; c < H; c += 256) { float v = xr[c]; s += v * v; }
  red[t] = s; __syncthreads();
  for (int o2 = 128; o2 > 0; o2 >>= 1) { if (t < o2) red[t] += red[t + o2]; __syncthreads(); }
  float scale = rsqrtf(red[0] / (float)H + 1e-5f);
  float* orow = o + (size_t)row * H;
  for (int c = t; c < H; c += 256) orow[c] = xr[c] * scale * w[c];
}

// ---------------- causal conv (K=4) + SiLU ----------------
__global__ void __launch_bounds__(256)
conv_silu_k(const float* __restrict__ xz, const float* __restrict__ cw,
            const float* __restrict__ cb, float* __restrict__ xb)
{
  int t  = blockIdx.x * 256 + threadIdx.x;           // < M*I
  int i  = t % I;
  int ml = t / I;                                    // b*L + l
  int l  = ml % L;
  float acc = cb[i];
  #pragma unroll
  for (int k = 0; k < Kc; ++k) {
    int ll = l - (Kc - 1) + k;
    if (ll >= 0) acc += xz[(size_t)(ml - (Kc - 1) + k) * (2 * I) + i] * cw[i * Kc + k];
  }
  xb[t] = acc / (1.f + __expf(-acc));                // silu
}

// ---------------- selective scan ----------------
// Block = 256 channels of one batch b; 16 SSM states per thread in VGPRs;
// per-timestep B/C vectors broadcast through LDS. grid = B * (I/256) = 24.
__global__ void __launch_bounds__(256)
scan_k(const float* __restrict__ dt, const float* __restrict__ xb,
       const float* __restrict__ xz, const float* __restrict__ proj,
       const float* __restrict__ A_log, const float* __restrict__ Dv,
       const float* __restrict__ s0, float* __restrict__ y, float* __restrict__ sT)
{
  __shared__ float Bs[S], Cs[S];
  const int b  = blockIdx.x / (I / 256);
  const int i  = (blockIdx.x % (I / 256)) * 256 + threadIdx.x;
  float A[S], st[S];
  #pragma unroll
  for (int s = 0; s < S; ++s) {
    A[s]  = -__expf(A_log[(size_t)i * S + s]);
    st[s] = s0 ? s0[((size_t)b * I + i) * S + s] : 0.f;
  }
  const float Di = Dv[i];
  for (int l = 0; l < L; ++l) {
    const int m = b * L + l;
    if (threadIdx.x < 32) {
      int s = threadIdx.x & 15;
      float v = proj[(size_t)m * PJ + ((threadIdx.x < 16) ? (R + s) : (R + S + s))];
      if (threadIdx.x < 16) Bs[s] = v; else Cs[s] = v;
    }
    __syncthreads();
    float dtv = dt[(size_t)m * I + i];
    float xv  = xb[(size_t)m * I + i];
    float acc = 0.f;
    #pragma unroll
    for (int s = 0; s < S; ++s) {
      float dA = __expf(dtv * A[s]);
      st[s] = dA * st[s] + dtv * Bs[s] * xv;
      acc  += st[s] * Cs[s];
    }
    float zv = xz[(size_t)m * (2 * I) + I + i];
    y[(size_t)m * I + i] = (acc + Di * xv) * (zv / (1.f + __expf(-zv)));
    __syncthreads();
  }
  #pragma unroll
  for (int s = 0; s < S; ++s) sT[((size_t)b * I + i) * S + s] = st[s];
}

// ---------------- states + noise, KL partial sums ----------------
__global__ void __launch_bounds__(256)
noise_kl_k(const float* __restrict__ encS, const float* __restrict__ noise,
           float* __restrict__ outS, float* __restrict__ part)
{
  __shared__ float red[256];
  int t = blockIdx.x * 256 + threadIdx.x;            // exact: NLBIS = 4608*256
  float v = encS[t];
  outS[t] = v + 0.5f * noise[t];
  red[threadIdx.x] = 0.5f * v * v;
  __syncthreads();
  for (int o = 128; o > 0; o >>= 1) { if (threadIdx.x < o) red[threadIdx.x] += red[threadIdx.x + o]; __syncthreads(); }
  if (threadIdx.x == 0) part[blockIdx.x] = red[0];
}

// ---------------- per-row log-softmax NLL ----------------
__global__ void __launch_bounds__(256)
rowloss_k(const float* __restrict__ logits, const int* __restrict__ ids,
          float* __restrict__ rowloss)
{
  __shared__ float red[256];
  int row = blockIdx.x;                              // 0..B*(L-1)-1
  int b = row / (L - 1), l = row % (L - 1);
  const float* lg = logits + (size_t)(b * L + l) * V;
  int t = threadIdx.x;
  float mx = -3.4e38f;
  for (int c = t; c < V; c += 256) mx = fmaxf(mx, lg[c]);
  red[t] = mx; __syncthreads();
  for (int o = 128; o > 0; o >>= 1) { if (t < o) red[t] = fmaxf(red[t], red[t + o]); __syncthreads(); }
  mx = red[0]; __syncthreads();
  float sm = 0.f;
  for (int c = t; c < V; c += 256) sm += __expf(lg[c] - mx);
  red[t] = sm; __syncthreads();
  for (int o = 128; o > 0; o >>= 1) { if (t < o) red[t] += red[t + o]; __syncthreads(); }
  if (t == 0) {
    int label = ids[b * L + l + 1];
    rowloss[row] = -(lg[label] - mx - logf(red[0]));
  }
}

// ---------------- final scalar reductions (deterministic) ----------------
__global__ void __launch_bounds__(256)
finalize_k(const float* __restrict__ rl, const float* __restrict__ klp, float* __restrict__ out)
{
  __shared__ float red[256];
  int t = threadIdx.x;
  float s = 0.f;
  for (int i2 = t; i2 < B * (L - 1); i2 += 256) s += rl[i2];
  red[t] = s; __syncthreads();
  for (int o = 128; o > 0; o >>= 1) { if (t < o) red[t] += red[t + o]; __syncthreads(); }
  if (t == 0) out[0] = red[0] / (float)(B * (L - 1));
  __syncthreads();
  s = 0.f;
  for (int i2 = t; i2 < NLBIS / 256; i2 += 256) s += klp[i2];
  red[t] = s; __syncthreads();
  for (int o = 128; o > 0; o >>= 1) { if (t < o) red[t] += red[t + o]; __syncthreads(); }
  if (t == 0) out[1] = red[0] / (float)B;
}

// =====================================================================
// Host orchestration
// =====================================================================
struct MParams {
  const float *emb, *norm, *in_proj, *conv_w, *conv_b, *x_proj,
              *dt_w, *dt_b, *A_log, *D, *out_proj;
};

template<int N, int K, int LDA, int LDB, int EPI>
static inline void gemm(const float* A, const float* W, const float* bias,
                        const float* Res, float* C, hipStream_t st)
{
  dim3 g((N + 255) / 256, M / 16);
  gemm_k<N, K, LDA, LDB, EPI><<<g, 256, 0, st>>>(A, W, bias, Res, C);
}

static void run_model(const int* ids, const MParams& P, const float* s0base,
                      float* sTbase, float* hA, float* hB, float* hn, float* xz,
                      float* xb, float* proj, float* dtb, float* yb,
                      hipStream_t st, float** hfinal)
{
  embed_k<<<M * H / 256, 256, 0, st>>>(ids, P.emb, hA);
  float* hc = hA; float* ho = hB;
  for (int ly = 0; ly < NL; ++ly) {
    const float* norm_l = P.norm     + (size_t)ly * H;
    const float* inp_l  = P.in_proj  + (size_t)ly * 2 * I * H;
    const float* cw_l   = P.conv_w   + (size_t)ly * I * Kc;
    const float* cb_l   = P.conv_b   + (size_t)ly * I;
    const float* xp_l   = P.x_proj   + (size_t)ly * PJ * I;
    const float* dtw_l  = P.dt_w     + (size_t)ly * I * R;
    const float* dtb_l  = P.dt_b     + (size_t)ly * I;
    const float* Al_l   = P.A_log    + (size_t)ly * I * S;
    const float* D_l    = P.D        + (size_t)ly * I;
    const float* op_l   = P.out_proj + (size_t)ly * H * I;

    rmsnorm_k<<<M, 256, 0, st>>>(hc, norm_l, hn);
    gemm<2 * I, H, H, H, 0>(hn, inp_l, nullptr, nullptr, xz, st);        // in_proj
    conv_silu_k<<<M * I / 256, 256, 0, st>>>(xz, cw_l, cb_l, xb);
    gemm<PJ, I, I, I, 0>(xb, xp_l, nullptr, nullptr, proj, st);          // x_proj
    gemm<I, R, PJ, R, 1>(proj, dtw_l, dtb_l, nullptr, dtb, st);          // dt + softplus
    scan_k<<<B * (I / 256), 256, 0, st>>>(dtb, xb, xz, proj, Al_l, D_l,
        s0base ? s0base + (size_t)ly * BIS : nullptr, yb,
        sTbase + (size_t)ly * BIS);
    gemm<H, I, I, I, 2>(yb, op_l, nullptr, hc, ho, st);                  // out_proj + residual
    float* tmp = hc; hc = ho; ho = tmp;
  }
  *hfinal = hc;
}

extern "C" void kernel_launch(void* const* d_in, const int* in_sizes, int n_in,
                              void* d_out, int out_size, void* d_ws, size_t ws_size,
                              hipStream_t stream)
{
  (void)in_sizes; (void)n_in; (void)out_size; (void)ws_size;

  const int*   ids   = (const int*)d_in[0];
  const float* noise = (const float*)d_in[1];
  MParams enc{ (const float*)d_in[2],  (const float*)d_in[3],  (const float*)d_in[4],
               (const float*)d_in[5],  (const float*)d_in[6],  (const float*)d_in[7],
               (const float*)d_in[8],  (const float*)d_in[9],  (const float*)d_in[10],
               (const float*)d_in[11], (const float*)d_in[12] };
  MParams dec{ (const float*)d_in[13], (const float*)d_in[14], (const float*)d_in[15],
               (const float*)d_in[16], (const float*)d_in[17], (const float*)d_in[18],
               (const float*)d_in[19], (const float*)d_in[20], (const float*)d_in[21],
               (const float*)d_in[22], (const float*)d_in[23] };
  const float* norm_f = (const float*)d_in[24];

  float* out       = (float*)d_out;
  float* logits    = out + 2;                        // B*L*V
  float* outStates = out + 2 + (size_t)M * V;        // NL*B*I*S (noised states)

  float* w    = (float*)d_ws;
  float* hA   = w; w += (size_t)M * H;
  float* hB   = w; w += (size_t)M * H;
  float* hn   = w; w += (size_t)M * H;
  float* xz   = w; w += (size_t)M * 2 * I;
  float* xb   = w; w += (size_t)M * I;
  float* proj = w; w += (size_t)M * PJ;
  float* dtb  = w; w += (size_t)M * I;
  float* yb   = w; w += (size_t)M * I;
  float* encS = w; w += (size_t)NLBIS;
  float* klp  = w; w += NLBIS / 256;
  float* rl   = w; w += B * (L - 1);

  float* hfin = nullptr;

  // encoder (zero init states), raw final states -> encS
  run_model(ids, enc, nullptr, encS, hA, hB, hn, xz, xb, proj, dtb, yb, stream, &hfin);

  // outStates = encS + 0.5*noise ; KL partials from raw encS
  noise_kl_k<<<NLBIS / 256, 256, 0, stream>>>(encS, noise, outStates, klp);

  // decoder seeded with noised states (sink final states into encS, unused)
  run_model(ids, dec, outStates, encS, hA, hB, hn, xz, xb, proj, dtb, yb, stream, &hfin);

  // final norm + vocab projection + loss
  rmsnorm_k<<<M, 256, 0, stream>>>(hfin, norm_f, hn);
  gemm<V, H, H, H, 0>(hn, dec.emb, nullptr, nullptr, logits, stream);
  rowloss_k<<<B * (L - 1), 256, 0, stream>>>(logits, ids, rl);
  finalize_k<<<1, 256, 0, stream>>>(rl, klp, out);
}
